// TPI_80831284511209
// MI455X (gfx1250) — matrix operand order repaired and checked by reading, not compiled, not run on hardware
//
#include <hip/hip_runtime.h>
#include <math.h>

#define DIMC 1024
#define BC   4
#define SC   4096
#define MC   (BC * SC)   // 16384 rows for every GEMM

typedef __attribute__((ext_vector_type(16))) _Float16 v16h;
typedef __attribute__((ext_vector_type(8)))  float    v8f;

// ---------------- f32 -> f16 convert (grid-stride) ----------------
__global__ __launch_bounds__(256) void cvt_f32_f16(const float* __restrict__ in,
                                                   _Float16* __restrict__ out, long n) {
  long i = (long)blockIdx.x * blockDim.x + threadIdx.x;
  long stride = (long)gridDim.x * blockDim.x;
  for (; i < n; i += stride) out[i] = (_Float16)in[i];
}

__device__ __forceinline__ float gelu_exact(float v) {
  return 0.5f * v * (1.0f + erff(v * 0.70710678118654752f));
}
__device__ __forceinline__ float sigm(float v) { return 1.0f / (1.0f + expf(-v)); }

// ---------------- WMMA GEMM: C = A(MxK,f16) * B(KxN,f16) + bias, epilogue ----
// Block tile 256(M) x 128(N), 8 waves (4x2), each wave 64x64 via 4x4 WMMA
// 16x16x32 tiles -> 16 WMMAs per 16 fragment ds_load_b128 per K-step.
// A tile staged with async global->LDS copies (ASYNCcnt); B tile staged with
// regular loads + transposed ds_store_b16 (needs the transpose anyway).
// EPI: 0 identity, 1 sigmoid, 2 sigmoid*5, 3 exact GELU
template<int EPI, bool OUT_HALF, bool RESID>
__global__ __launch_bounds__(256) void gemm_wmma_f16(
    const _Float16* __restrict__ A, const _Float16* __restrict__ Bm,
    const float* __restrict__ bias, void* __restrict__ Cout,
    const float* __restrict__ Res, int M, int N, int K)
{
  __shared__ __align__(16) _Float16 As[256 * 32];  // [row][k]   16 KB
  __shared__ __align__(16) _Float16 Bt[128 * 32];  // [col][k]    8 KB

  const int tid  = threadIdx.x;
  const int lane = tid & 31;
  const int wid  = tid >> 5;     // 0..7
  const int wm   = wid >> 1;     // 0..3  (M sub-tile, 64 rows each)
  const int wn   = wid & 1;      // 0..1  (N sub-tile, 64 cols each)
  const int m0   = blockIdx.x * 256;
  const int n0   = blockIdx.y * 128;

  v8f acc[4][4] = {};

  for (int kk = 0; kk < K; kk += 32) {
    // ---- stage A tile 256x32 via async global->LDS (4 x b128 per thread) ----
    #pragma unroll
    for (int rep = 0; rep < 4; ++rep) {
      int i = tid + rep * 256;              // 0..1023 16-byte chunks
      int row = i >> 2, c8 = (i & 3) * 8;
      const _Float16* gp = A + (size_t)(m0 + row) * K + kk + c8;
      unsigned ldst = (unsigned)(uintptr_t)&As[(size_t)i * 8];
      asm volatile("global_load_async_to_lds_b128 %0, %1, off"
                   :: "v"(ldst), "v"(gp) : "memory");
    }
    // ---- stage B tile 32x128, transposed into Bt[col][k] ----
    #pragma unroll
    for (int rep = 0; rep < 2; ++rep) {
      int t2 = tid + rep * 256;             // 0..511 8-half chunks
      int k = t2 >> 4, c8 = (t2 & 15) * 8;
      union { uint4 v; _Float16 h[8]; } u;
      u.v = *(const uint4*)(Bm + (size_t)(kk + k) * N + n0 + c8);
      #pragma unroll
      for (int j = 0; j < 8; ++j) Bt[(c8 + j) * 32 + k] = u.h[j];
    }
    if (kk + 32 < K)  // pull next B K-slice toward L2 (global_prefetch_b8)
      __builtin_prefetch(Bm + (size_t)(kk + 32 + (tid >> 4)) * N + n0, 0, 1);
    asm volatile("s_wait_asynccnt 0" ::: "memory");
    __syncthreads();

    // ---- build fragments and issue 4x4 WMMAs ----
    v16h af[4], bf[4];
    const int lr  = lane & 15;
    const int khA = (lane < 16) ? 0 : 8;   // A: halves {kh..kh+7, kh+16..kh+23}
    #pragma unroll
    for (int i = 0; i < 4; ++i) {
      int rb = (wm * 64 + i * 16 + lr) * 32;
      ((uint4*)&af[i])[0] = *(const uint4*)&As[rb + khA];
      ((uint4*)&af[i])[1] = *(const uint4*)&As[rb + khA + 16];
    }
    const int khB = (lane < 16) ? 0 : 8;   // B: halves {kh..kh+7, kh+16..kh+23}
    #pragma unroll
    for (int j = 0; j < 4; ++j) {
      int cb = (wn * 64 + j * 16 + lr) * 32 + khB;
      ((uint4*)&bf[j])[0] = *(const uint4*)&Bt[cb];
      ((uint4*)&bf[j])[1] = *(const uint4*)&Bt[cb + 16];
    }
    #pragma unroll
    for (int i = 0; i < 4; ++i)
      #pragma unroll
      for (int j = 0; j < 4; ++j)
        acc[i][j] = __builtin_amdgcn_wmma_f32_16x16x32_f16(
            false, af[i], false, bf[j], (short)0, acc[i][j], false, false);
    __syncthreads();
  }

  // ---- epilogue: C layout VGPR r -> M = r + (lane<16?0:8), N = lane&15 ----
  const int lr = lane & 15;
  const int lm = (lane < 16) ? 0 : 8;
  #pragma unroll
  for (int i = 0; i < 4; ++i) {
    #pragma unroll
    for (int j = 0; j < 4; ++j) {
      int colg = n0 + wn * 64 + j * 16 + lr;
      float bv = bias[colg];
      #pragma unroll
      for (int r = 0; r < 8; ++r) {
        int rowg = m0 + wm * 64 + i * 16 + lm + r;
        float v = acc[i][j][r] + bv;
        if (EPI == 1)      v = sigm(v);
        else if (EPI == 2) v = 5.0f * sigm(v);
        else if (EPI == 3) v = gelu_exact(v);
        size_t idx = (size_t)rowg * N + colg;
        if (RESID) v += Res[idx];
        if (OUT_HALF) ((_Float16*)Cout)[idx] = (_Float16)v;
        else          ((float*)Cout)[idx]    = v;
      }
    }
  }
}

// ---------------- sequential phase scan over S, one thread per (b,d) --------
__global__ __launch_bounds__(256) void phase_scan(
    const float* __restrict__ x, const float* __restrict__ omega,
    const float* __restrict__ mag, const float* __restrict__ gate,
    const float* __restrict__ qoff, const float* __restrict__ phiinit,
    const float* __restrict__ int_scale, float* __restrict__ ctx)
{
  int t = blockIdx.x * blockDim.x + threadIdx.x;   // 0..B*DIM-1
  int b = t >> 10;
  int d = t & (DIMC - 1);
  float isc = fabsf(int_scale[d]);
  float accP = 0.f, accR = 0.f, accI = 0.f, accM = 0.f;
  size_t base  = (size_t)b * SC * DIMC + d;
  size_t cbase = (size_t)b * SC * (4 * DIMC) + d;
  for (int s = 0; s < SC; ++s) {
    size_t o = base + (size_t)s * DIMC;
    float xv = x[o], om = omega[o], mg = mag[o];
    float gt = gate[o], qo = qoff[o], p0 = phiinit[o];
    accP += gt * (om * isc);
    float phi = p0 + accP;
    float sp, cp; sincosf(phi, &sp, &cp);
    float wc = mg * xv;
    accR += wc * cp; accI += wc * sp; accM += mg;
    float rs = rsqrtf(accM + 1e-8f);
    float mr = accR * rs, mi = accI * rs;
    float sq, cq; sincosf(phi + qo, &sq, &cq);
    size_t co = cbase + (size_t)s * (4 * DIMC);
    ctx[co]            = xv * cp;
    ctx[co + DIMC]     = xv * sp;
    ctx[co + 2 * DIMC] = mr * cq + mi * sq;
    ctx[co + 3 * DIMC] = mi * cq - mr * sq;
  }
}

// ---------------- LayerNorm over 4096, one block per row, f16 out ----------
__global__ __launch_bounds__(256) void ln_rows(
    const float* __restrict__ ctx, const float* __restrict__ g,
    const float* __restrict__ b, _Float16* __restrict__ out)
{
  const int C = 4 * DIMC;
  size_t row = blockIdx.x;
  const float* p = ctx + row * C;
  float s = 0.f, s2 = 0.f;
  for (int i = threadIdx.x; i < C; i += 256) { float v = p[i]; s += v; s2 += v * v; }
  __shared__ float r1[256], r2[256];
  r1[threadIdx.x] = s; r2[threadIdx.x] = s2;
  __syncthreads();
  for (int st = 128; st > 0; st >>= 1) {
    if (threadIdx.x < st) { r1[threadIdx.x] += r1[threadIdx.x + st];
                            r2[threadIdx.x] += r2[threadIdx.x + st]; }
    __syncthreads();
  }
  float mu  = r1[0] / C;
  float var = r2[0] / C - mu * mu;
  float rsv = rsqrtf(var + 1e-5f);
  _Float16* o = out + row * C;
  for (int i = threadIdx.x; i < C; i += 256)
    o[i] = (_Float16)((p[i] - mu) * rsv * g[i] + b[i]);
}

// ============================ host launcher =================================
extern "C" void kernel_launch(void* const* d_in, const int* in_sizes, int n_in,
                              void* d_out, int out_size, void* d_ws, size_t ws_size,
                              hipStream_t stream) {
  (void)in_sizes; (void)n_in; (void)out_size; (void)ws_size;
  const float* x        = (const float*)d_in[0];
  const float* W_omega  = (const float*)d_in[1];
  const float* b_omega  = (const float*)d_in[2];
  const float* W_p1     = (const float*)d_in[3];
  const float* b_p1     = (const float*)d_in[4];
  const float* W_p2     = (const float*)d_in[5];
  const float* b_p2     = (const float*)d_in[6];
  const float* W_gate   = (const float*)d_in[7];
  const float* b_gate   = (const float*)d_in[8];
  const float* W_mag    = (const float*)d_in[9];
  const float* b_mag    = (const float*)d_in[10];
  const float* W_qoff   = (const float*)d_in[11];
  const float* b_qoff   = (const float*)d_in[12];
  const float* intscale = (const float*)d_in[13];
  const float* ln_g     = (const float*)d_in[14];
  const float* ln_b     = (const float*)d_in[15];
  const float* W_o1     = (const float*)d_in[16];
  const float* b_o1     = (const float*)d_in[17];
  const float* W_o2     = (const float*)d_in[18];
  const float* b_o2     = (const float*)d_in[19];
  float* out = (float*)d_out;

  char* ws = (char*)d_ws;
  size_t cur = 0;
  auto carve = [&](size_t bytes) -> void* {
    void* p = (void*)(ws + cur);
    cur += (bytes + 255) & ~(size_t)255;
    return p;
  };

  const size_t MD = (size_t)MC * DIMC;               // 16M elems
  _Float16* x16    = (_Float16*)carve(MD * 2);
  _Float16* w16om  = (_Float16*)carve((size_t)DIMC * DIMC * 2);
  _Float16* w16p1  = (_Float16*)carve((size_t)DIMC * DIMC * 2);
  _Float16* w16p2  = (_Float16*)carve((size_t)DIMC * DIMC * 2);
  _Float16* w16gt  = (_Float16*)carve((size_t)DIMC * DIMC * 2);
  _Float16* w16mg  = (_Float16*)carve((size_t)DIMC * DIMC * 2);
  _Float16* w16qo  = (_Float16*)carve((size_t)DIMC * DIMC * 2);
  _Float16* w16o1  = (_Float16*)carve((size_t)(4 * DIMC) * (2 * DIMC) * 2);
  _Float16* w16o2  = (_Float16*)carve((size_t)(2 * DIMC) * DIMC * 2);
  float* omega   = (float*)carve(MD * 4);            // 5x64MB region (dead after scan)
  float* mag     = (float*)carve(MD * 4);
  float* gate    = (float*)carve(MD * 4);
  float* qoff    = (float*)carve(MD * 4);
  float* phiinit = (float*)carve(MD * 4);
  _Float16* p1a16 = (_Float16*)carve(MD * 2);
  float* ctx      = (float*)carve(MD * 4 * 4);       // [M][4096] f32
  // reuse the dead fp32 GEMM-output region for late-stage f16 tensors
  _Float16* ctxn16 = (_Float16*)omega;                         // 128 MB
  _Float16* h16    = (_Float16*)((char*)omega + MD * 4 * 2);   // 64 MB, disjoint

  // ---- precision conversion ----
  cvt_f32_f16<<<2048, 256, 0, stream>>>(x,       x16,   (long)MD);
  cvt_f32_f16<<<1024, 256, 0, stream>>>(W_omega, w16om, (long)DIMC * DIMC);
  cvt_f32_f16<<<1024, 256, 0, stream>>>(W_p1,    w16p1, (long)DIMC * DIMC);
  cvt_f32_f16<<<1024, 256, 0, stream>>>(W_p2,    w16p2, (long)DIMC * DIMC);
  cvt_f32_f16<<<1024, 256, 0, stream>>>(W_gate,  w16gt, (long)DIMC * DIMC);
  cvt_f32_f16<<<1024, 256, 0, stream>>>(W_mag,   w16mg, (long)DIMC * DIMC);
  cvt_f32_f16<<<1024, 256, 0, stream>>>(W_qoff,  w16qo, (long)DIMC * DIMC);
  cvt_f32_f16<<<2048, 256, 0, stream>>>(W_o1,    w16o1, (long)(4 * DIMC) * (2 * DIMC));
  cvt_f32_f16<<<1024, 256, 0, stream>>>(W_o2,    w16o2, (long)(2 * DIMC) * DIMC);

  // ---- six input-projection GEMMs with fused activations ----
  dim3 gA(MC / 256, DIMC / 128);
  gemm_wmma_f16<0, false, false><<<gA, 256, 0, stream>>>(x16, w16om, b_omega, omega,  nullptr, MC, DIMC, DIMC);
  gemm_wmma_f16<2, false, false><<<gA, 256, 0, stream>>>(x16, w16mg, b_mag,   mag,    nullptr, MC, DIMC, DIMC);
  gemm_wmma_f16<1, false, false><<<gA, 256, 0, stream>>>(x16, w16gt, b_gate,  gate,   nullptr, MC, DIMC, DIMC);
  gemm_wmma_f16<0, false, false><<<gA, 256, 0, stream>>>(x16, w16qo, b_qoff,  qoff,   nullptr, MC, DIMC, DIMC);
  gemm_wmma_f16<3, true,  false><<<gA, 256, 0, stream>>>(x16, w16p1, b_p1,    p1a16,  nullptr, MC, DIMC, DIMC);
  gemm_wmma_f16<0, false, false><<<gA, 256, 0, stream>>>(p1a16, w16p2, b_p2,  phiinit,nullptr, MC, DIMC, DIMC);

  // ---- cumulative phase / complex memory scan -> context [M][4096] ----
  phase_scan<<<(BC * DIMC) / 256, 256, 0, stream>>>(x, omega, mag, gate, qoff,
                                                    phiinit, intscale, ctx);

  // ---- layernorm -> f16 normalized context ----
  ln_rows<<<MC, 256, 0, stream>>>(ctx, ln_g, ln_b, ctxn16);

  // ---- output MLP: 4096->2048 (GELU) then 2048->1024 (+residual x) ----
  dim3 gO1(MC / 256, (2 * DIMC) / 128);
  gemm_wmma_f16<3, true,  false><<<gO1, 256, 0, stream>>>(ctxn16, w16o1, b_o1, h16, nullptr, MC, 2 * DIMC, 4 * DIMC);
  dim3 gO2(MC / 256, DIMC / 128);
  gemm_wmma_f16<0, false, true ><<<gO2, 256, 0, stream>>>(h16, w16o2, b_o2, out, x, MC, DIMC, 2 * DIMC);
}
